// CSPLayer_43473658970185
// MI455X (gfx1250) — matrix operand order repaired and checked by reading, not compiled, not run on hardware
//
#include <hip/hip_runtime.h>

// ---------------------------------------------------------------------------
// CDNA5 / gfx1250 implementation of the CSP message-passing layer.
// - All GEMMs on v_wmma_f32_16x16x32_f16 (f16 in, f32 accumulate).
// - Weights pre-permuted once into workspace (WMMA B-tile lane layout),
//   then DMA'd into LDS per block with the Tensor Data Mover
//   (tensor_load_to_lds + s_wait_tensorcnt).
// - Segment sums via f32 global atomics (L2-resident).
// ---------------------------------------------------------------------------

typedef __attribute__((ext_vector_type(16))) _Float16     v16h;
typedef __attribute__((ext_vector_type(8)))  float        v8f;
typedef __attribute__((ext_vector_type(4)))  unsigned int u32x4;
typedef __attribute__((ext_vector_type(8)))  int          i32x8;
typedef __attribute__((ext_vector_type(4)))  int          i32x4;

union V16 { v16h h; uint4 u[2]; };

__device__ __forceinline__ v8f wmma16(v16h a, v16h b, v8f c) {
  // 8 args: (neg_a, A, neg_b, B, c_mod, C, reuse_a, reuse_b)
  return __builtin_amdgcn_wmma_f32_16x16x32_f16(false, a, false, b, (short)0, c,
                                                false, false);
}

__device__ __forceinline__ float silu_f(float v) {
  return v / (1.0f + __expf(-v));
}

// ---------------------------------------------------------------------------
// Tensor Data Mover: 1-D contiguous copy global -> LDS (data_size = 8B).
// D# built per CDNA5 ISA ch.8 (group0: flags/lds/global/type, group1: dims).
// This toolchain exposes the 6-arg builtin:
//   (u32x4 g0, i32x8 g1, i32x4 g2, i32x4 g3, i32x8 extra, i32 cpol)
// ---------------------------------------------------------------------------
__device__ __forceinline__ void tdm_copy_to_lds(const void* gsrc,
                                                unsigned lds_off,
                                                unsigned nbytes) {
  unsigned long long ga = (unsigned long long)gsrc;
  unsigned n8 = nbytes >> 3;  // elements of 8 bytes (tile_dim0, must be <64K)

  u32x4 g0;
  g0[0] = 1u;                                   // count=1, user mode, no gather
  g0[1] = lds_off;                              // lds_addr (bytes)
  g0[2] = (unsigned)(ga & 0xFFFFFFFFu);         // global_addr[31:0]
  g0[3] = (unsigned)(((ga >> 32) & 0x1FFFFFFu)  // global_addr[56:32]
                     | (2u << 30));             // type = 2 ("image")

  i32x8 g1;
  g1[0] = (int)(3u << 16);                      // wg_mask=0, data_size=8B
  g1[1] = (int)((n8 & 0xFFFFu) << 16);          // tensor_dim0[15:0]
  g1[2] = (int)((n8 >> 16) | (1u << 16));       // tensor_dim0[31:16], tensor_dim1=1
  g1[3] = (int)((n8 & 0xFFFFu) << 16);          // tensor_dim1 hi=0, tile_dim0=n8
  g1[4] = (int)(1u | (1u << 16));               // tile_dim1=1, tile_dim2=1
  g1[5] = (int)n8;                              // tensor_dim0_stride[31:0]
  g1[6] = 0;                                    // stride hi, tensor_dim1_stride lo
  g1[7] = 0;

  i32x4 g2; g2[0] = 1; g2[1] = 1; g2[2] = 0; g2[3] = 0;   // tensor_dim2/3 = 1
  i32x4 g3; g3[0] = 0; g3[1] = (int)(1u << 16); g3[2] = 0; g3[3] = 0; // dim4=1

  i32x8 gx;
#pragma unroll
  for (int i = 0; i < 8; ++i) gx[i] = 0;

  __builtin_amdgcn_tensor_load_to_lds(g0, g1, g2, g3, gx, 0);
}

// ---- B tile from LDS, stored pre-permuted: (((kt*8+nt)*32+lane)*16) --------
__device__ __forceinline__ v16h ldsB(const _Float16* W, int kt, int nt, int lane) {
  const uint4* p = (const uint4*)(W + ((((kt << 3) + nt) << 5) + lane) * 16);
  V16 t; t.u[0] = p[0]; t.u[1] = p[1];
  return t.h;
}

// ---- A tile from per-wave LDS staging, stored pre-permuted -----------------
__device__ __forceinline__ v16h ldsA(const _Float16* S, int kt, int lane) {
  const uint4* p = (const uint4*)(S + (((kt << 5) + lane) << 4));
  V16 t; t.u[0] = p[0]; t.u[1] = p[1];
  return t.h;
}

// ---- A tile directly from a global f16 row (row = lane&15's node) ----------
// ISA A layout: lanes 0-15 hold K = {0..7, 16..23}, lanes 16-31 K = {8..15, 24..31}
// -> two contiguous 16B runs per lane.
__device__ __forceinline__ v16h gA_f16(const _Float16* rowBase, int lane) {
  int k0 = (lane >> 4) << 3;
  const uint4* p = (const uint4*)(rowBase + k0);
  V16 t; t.u[0] = p[0]; t.u[1] = p[2];  // +16 halves = +32B = 2 uint4
  return t.h;
}

// ---- A tile from a global f32 row (converted to f16 on the fly) ------------
__device__ __forceinline__ v16h gA_f32(const float* rowBase, int lane) {
  int k0 = (lane >> 4) << 3;
  V16 t;
#pragma unroll
  for (int i = 0; i < 8; ++i) {
    t.h[i]     = (_Float16)rowBase[k0 + i];
    t.h[i + 8] = (_Float16)rowBase[k0 + 16 + i];
  }
  return t.h;
}

// ---- A tile from edge_attr (20 real cols, zero-padded to K=32) -------------
__device__ __forceinline__ v16h gA_edge(const float* ea, int row, int lane) {
  int k0 = (lane >> 4) << 3;
  const float* r = ea + row * 20;
  V16 t;
#pragma unroll
  for (int i = 0; i < 8; ++i) {
    int k1 = k0 + i;
    int k2 = k0 + 16 + i;
    t.h[i]     = (k1 < 20) ? (_Float16)r[k1] : (_Float16)0.0f;
    t.h[i + 8] = (k2 < 20) ? (_Float16)r[k2] : (_Float16)0.0f;
  }
  return t.h;
}

// ---- scatter one C tile (f32 per-lane vals) into A-layout staging ----------
// C layout: value (m = r + 8*half, n = jt*16 + (lane&15)).  n becomes k.
__device__ __forceinline__ void stageC(_Float16* S, int jt, int lane,
                                       const float* vals) {
  int lm = lane & 15, half = lane >> 4;
  int k  = (jt << 4) + lm;
  int kt = k >> 5;
  int hb = (k >> 3) & 1;
  int i  = (k & 7) + (((k >> 4) & 1) << 3);
#pragma unroll
  for (int r = 0; r < 8; ++r) {
    int m     = r + (half << 3);
    int laneP = m + (hb << 4);
    S[(((kt << 5) + laneP) << 4) + i] = (_Float16)vals[r];
  }
}

// ===========================================================================
// Weight prep: convert f32 (Kx128) -> f16 in WMMA B-tile lane layout, packed
// contiguously in workspace:
//   [0]      ee_w1 (pad K=32)   4096
//   [4096]   ee_w2 (K=128)     16384
//   [20480]  mn_w1 (K=384)     49152
//   [69632]  mn_w2 (K=128)     16384   -- edge block: 86016 f16 = 172032 B
//   [86016]  un_w1 (K=256)     32768
//   [118784] un_w2 (K=128)     16384   -- node block: 49152 f16 =  98304 B
// ===========================================================================
__device__ __forceinline__ void permW(_Float16* dst, const float* W, int Kreal,
                                      int p) {
  int elem = p & 15;
  int lane = (p >> 4) & 31;
  int nt   = (p >> 9) & 7;
  int kt   = p >> 12;
  int n = (nt << 4) + (lane & 15);
  int k = (kt << 5) + ((lane >> 4) << 3) + (elem & 7) + ((elem >> 3) << 4);
  float v = (k < Kreal) ? W[k * 128 + n] : 0.0f;
  dst[p] = (_Float16)v;
}

__global__ void wprep_kernel(const float* ee_w1, const float* ee_w2,
                             const float* mn_w1, const float* mn_w2,
                             const float* un_w1, const float* un_w2,
                             _Float16* __restrict__ wp) {
  int t = blockIdx.x * blockDim.x + threadIdx.x;
  if (t < 4096)        permW(wp,          ee_w1,  20, t);
  else if (t < 20480)  permW(wp + 4096,   ee_w2, 128, t - 4096);
  else if (t < 69632)  permW(wp + 20480,  mn_w1, 384, t - 20480);
  else if (t < 86016)  permW(wp + 69632,  mn_w2, 128, t - 69632);
  else if (t < 118784) permW(wp + 86016,  un_w1, 256, t - 86016);
  else if (t < 135168) permW(wp + 118784, un_w2, 128, t - 118784);
}

// ===========================================================================
// Kernel: x -> f16 copy, zero aggregation buffer
// ===========================================================================
__global__ void cvt_kernel(const float* __restrict__ x, _Float16* __restrict__ xh,
                           float* __restrict__ aggr, int total) {
  int i = blockIdx.x * blockDim.x + threadIdx.x;
  if (i < total) {
    xh[i]   = (_Float16)x[i];
    aggr[i] = 0.0f;
  }
}

// ===========================================================================
// Kernel: lattice_info = lattice_flat @ l2a_w + l2a_b ; zero sums/counts
// ===========================================================================
__global__ void latinfo_kernel(const float* __restrict__ lattice,
                               const float* __restrict__ w,
                               const float* __restrict__ b,
                               float* __restrict__ latinfo,
                               float* __restrict__ sums,
                               float* __restrict__ counts, int B) {
  int t = blockIdx.x * blockDim.x + threadIdx.x;
  if (t < B * 128) {
    int bb = t >> 7, h = t & 127;
    float s = b[h];
#pragma unroll
    for (int i = 0; i < 9; ++i) s += lattice[bb * 9 + i] * w[i * 128 + h];
    latinfo[t] = s;
    sums[t]    = 0.0f;
  }
  if (t < B) counts[t] = 0.0f;
}

// ===========================================================================
// Kernel: per-batch node counts
// ===========================================================================
__global__ void count_kernel(const int* __restrict__ batch,
                             float* __restrict__ counts, int N) {
  int i = blockIdx.x * blockDim.x + threadIdx.x;
  if (i < N) atomicAdd(&counts[batch[i]], 1.0f);
}

// ===========================================================================
// Edge pipeline.  Per 16-edge tile (one wave):
//   ef  = ee_w2 @ silu(ee_w1 @ pad(edge_attr))        (8 + 32 wmma)
//   msg = mn_w2 @ silu(mn_w1 @ [x[dst]|x[src]|ef])    (96 + 32 wmma)
//   aggr[dst] += msg                                  (f32 atomics, L2)
// Weights arrive via one TDM DMA (172KB); 4KB staging per wave.
// ===========================================================================
__global__ __launch_bounds__(256) void edge_kernel(
    const float* __restrict__ edge_attr, const int* __restrict__ eidx,
    const _Float16* __restrict__ xh, const _Float16* __restrict__ wperm,
    const float* ee_b1, const float* ee_b2, const float* mn_b1,
    const float* mn_b2, float* __restrict__ aggr, int E) {
  extern __shared__ _Float16 smem[];
  _Float16* W1 = smem;                  //  32x128 =  4096 f16
  _Float16* W2 = W1 + 4096;             // 128x128 = 16384
  _Float16* W3 = W2 + 16384;            // 384x128 = 49152
  _Float16* W4 = W3 + 49152;            // 128x128 = 16384
  _Float16* stageAll = W4 + 16384;      // 8 waves * 2048

  if (threadIdx.x < 32) {               // wave 0 issues the weight DMA
    tdm_copy_to_lds(wperm, (unsigned)(unsigned long long)(void*)smem,
                    86016u * 2u);
    __builtin_amdgcn_s_wait_tensorcnt(0);
  }
  __syncthreads();

  const int lane = threadIdx.x & 31;
  const int wave = threadIdx.x >> 5;
  const int lm   = lane & 15;
  const int half = lane >> 4;
  _Float16* stage = stageAll + wave * 2048;

  const int* srcI = eidx;
  const int* dstI = eidx + E;
  const int nTiles = E >> 4;
  const v8f zero = {0.f, 0.f, 0.f, 0.f, 0.f, 0.f, 0.f, 0.f};

  for (int tile = blockIdx.x * 8 + wave; tile < nTiles; tile += gridDim.x * 8) {
    const int eBase = tile << 4;
    const int myDst = dstI[eBase + lm];   // node for A-row lm
    const int mySrc = srcI[eBase + lm];

    // ---- edge encoder layer 1 (K=32, padded) ----
    {
      v16h aE = gA_edge(edge_attr, eBase + lm, lane);
#pragma unroll
      for (int jt = 0; jt < 8; ++jt) {
        v8f acc = wmma16(aE, ldsB(W1, 0, jt, lane), zero);
        float bias = ee_b1[(jt << 4) + lm];
        float vals[8];
#pragma unroll
        for (int r = 0; r < 8; ++r) vals[r] = silu_f(acc[r] + bias);
        stageC(stage, jt, lane, vals);
      }
    }

    // ---- edge encoder layer 2 (K=128) -> edge_feat ----
    {
      v8f acc[8];
#pragma unroll
      for (int jt = 0; jt < 8; ++jt) acc[jt] = zero;
#pragma unroll
      for (int kt = 0; kt < 4; ++kt) {
        v16h a = ldsA(stage, kt, lane);
#pragma unroll
        for (int jt = 0; jt < 8; ++jt)
          acc[jt] = wmma16(a, ldsB(W2, kt, jt, lane), acc[jt]);
      }
#pragma unroll
      for (int jt = 0; jt < 8; ++jt) {
        float bias = ee_b2[(jt << 4) + lm];
        float vals[8];
#pragma unroll
        for (int r = 0; r < 8; ++r) vals[r] = acc[jt][r] + bias;
        stageC(stage, jt, lane, vals);   // edge_feat -> staging
      }
    }

    // ---- message net layer 1 (K = 384 = x[dst] | x[src] | edge_feat) ----
    v8f acc4[8];
    {
      v8f acc[8];
#pragma unroll
      for (int jt = 0; jt < 8; ++jt) acc[jt] = zero;
      const _Float16* xd = xh + (long long)myDst * 128;
      const _Float16* xs = xh + (long long)mySrc * 128;
#pragma unroll
      for (int kt = 0; kt < 12; ++kt) {
        v16h a;
        if (kt < 4)      a = gA_f16(xd + (kt << 5), lane);
        else if (kt < 8) a = gA_f16(xs + ((kt - 4) << 5), lane);
        else             a = ldsA(stage, kt - 8, lane);
#pragma unroll
        for (int jt = 0; jt < 8; ++jt)
          acc[jt] = wmma16(a, ldsB(W3, kt, jt, lane), acc[jt]);
      }
#pragma unroll
      for (int jt = 0; jt < 8; ++jt) {
        float bias = mn_b1[(jt << 4) + lm];
        float vals[8];
#pragma unroll
        for (int r = 0; r < 8; ++r) vals[r] = silu_f(acc[jt][r] + bias);
        stageC(stage, jt, lane, vals);
      }

      // ---- message net layer 2 (K=128) ----
#pragma unroll
      for (int jt = 0; jt < 8; ++jt) acc4[jt] = zero;
#pragma unroll
      for (int kt = 0; kt < 4; ++kt) {
        v16h a = ldsA(stage, kt, lane);
#pragma unroll
        for (int jt = 0; jt < 8; ++jt)
          acc4[jt] = wmma16(a, ldsB(W4, kt, jt, lane), acc4[jt]);
      }
    }

    // ---- scatter: aggr[dst[m]] += msg[m]  (C layout: m = r + 8*half) ----
#pragma unroll
    for (int jt = 0; jt < 8; ++jt) {
      float bias = mn_b2[(jt << 4) + lm];
#pragma unroll
      for (int r = 0; r < 8; ++r) {
        int m    = r + (half << 3);
        int node = __shfl(myDst, m, 32);
        atomicAdd(&aggr[(long long)node * 128 + (jt << 4) + lm],
                  acc4[jt][r] + bias);
      }
    }
  }
}

// ===========================================================================
// Node update net + residual + lattice_info, accumulate batch sums
// ===========================================================================
__global__ __launch_bounds__(256) void node_kernel(
    const float* __restrict__ x, const _Float16* __restrict__ xh,
    const float* __restrict__ aggr, const _Float16* __restrict__ wperm_un,
    const float* un_b1, const float* un_b2,
    const float* __restrict__ latinfo, const int* __restrict__ batch,
    float* __restrict__ x2, float* __restrict__ sums, int N) {
  extern __shared__ _Float16 smem[];
  _Float16* W1 = smem;                  // 256x128 = 32768 f16
  _Float16* W2 = W1 + 32768;            // 128x128 = 16384
  _Float16* stageAll = W2 + 16384;      // 8 * 2048

  if (threadIdx.x < 32) {
    tdm_copy_to_lds(wperm_un, (unsigned)(unsigned long long)(void*)smem,
                    49152u * 2u);
    __builtin_amdgcn_s_wait_tensorcnt(0);
  }
  __syncthreads();

  const int lane = threadIdx.x & 31;
  const int wave = threadIdx.x >> 5;
  const int lm   = lane & 15;
  const int half = lane >> 4;
  _Float16* stage = stageAll + wave * 2048;

  const int nTiles = N >> 4;
  const v8f zero = {0.f, 0.f, 0.f, 0.f, 0.f, 0.f, 0.f, 0.f};

  for (int tile = blockIdx.x * 8 + wave; tile < nTiles; tile += gridDim.x * 8) {
    const int nodeRow = (tile << 4) + lm;
    const int myBatch = batch[nodeRow];
    const _Float16* xr = xh + (long long)nodeRow * 128;
    const float*    ar = aggr + (long long)nodeRow * 128;

    // ---- update net layer 1 (K = 256 = x | aggr) ----
    v8f acc[8];
#pragma unroll
    for (int jt = 0; jt < 8; ++jt) acc[jt] = zero;
#pragma unroll
    for (int kt = 0; kt < 8; ++kt) {
      v16h a = (kt < 4) ? gA_f16(xr + (kt << 5), lane)
                        : gA_f32(ar + ((kt - 4) << 5), lane);
#pragma unroll
      for (int jt = 0; jt < 8; ++jt)
        acc[jt] = wmma16(a, ldsB(W1, kt, jt, lane), acc[jt]);
    }
#pragma unroll
    for (int jt = 0; jt < 8; ++jt) {
      float bias = un_b1[(jt << 4) + lm];
      float vals[8];
#pragma unroll
      for (int r = 0; r < 8; ++r) vals[r] = silu_f(acc[jt][r] + bias);
      stageC(stage, jt, lane, vals);
    }

    // ---- update net layer 2 (K=128) ----
    v8f acc2[8];
#pragma unroll
    for (int jt = 0; jt < 8; ++jt) acc2[jt] = zero;
#pragma unroll
    for (int kt = 0; kt < 4; ++kt) {
      v16h a = ldsA(stage, kt, lane);
#pragma unroll
      for (int jt = 0; jt < 8; ++jt)
        acc2[jt] = wmma16(a, ldsB(W2, kt, jt, lane), acc2[jt]);
    }

    // ---- x2 = x + upd + lattice_info[batch];  sums[batch] += x2 ----
#pragma unroll
    for (int jt = 0; jt < 8; ++jt) {
      int n = (jt << 4) + lm;
      float bias = un_b2[n];
#pragma unroll
      for (int r = 0; r < 8; ++r) {
        int m    = r + (half << 3);
        int node = (tile << 4) + m;
        int b    = __shfl(myBatch, m, 32);
        float v  = acc2[jt][r] + bias + x[(long long)node * 128 + n] +
                   latinfo[(long long)b * 128 + n];
        x2[(long long)node * 128 + n] = v;
        atomicAdd(&sums[(long long)b * 128 + n], v);
      }
    }
  }
}

// ===========================================================================
// LayerNorm over each node row (wave per row) -> d_out
// ===========================================================================
__global__ __launch_bounds__(256) void node_ln_kernel(
    const float* __restrict__ x2, const float* __restrict__ g,
    const float* __restrict__ b, float* __restrict__ out, int N) {
  int row = blockIdx.x * 8 + (threadIdx.x >> 5);
  if (row >= N) return;
  int lane = threadIdx.x & 31;
  const float4* r = (const float4*)(x2 + (long long)row * 128);
  float4 v = r[lane];
  float s  = v.x + v.y + v.z + v.w;
  float sq = v.x * v.x + v.y * v.y + v.z * v.z + v.w * v.w;
#pragma unroll
  for (int off = 16; off > 0; off >>= 1) {
    s  += __shfl_xor(s, off, 32);
    sq += __shfl_xor(sq, off, 32);
  }
  float mu  = s * (1.0f / 128.0f);
  float var = sq * (1.0f / 128.0f) - mu * mu;
  float rs  = rsqrtf(var + 1e-5f);
  int c = lane << 2;
  float4 o;
  o.x = (v.x - mu) * rs * g[c + 0] + b[c + 0];
  o.y = (v.y - mu) * rs * g[c + 1] + b[c + 1];
  o.z = (v.z - mu) * rs * g[c + 2] + b[c + 2];
  o.w = (v.w - mu) * rs * g[c + 3] + b[c + 3];
  ((float4*)(out + (long long)row * 128))[lane] = o;
}

// ===========================================================================
// atom_info = sums/counts; lattice' = LN(lattice + atom_info @ a2l + b)
// ===========================================================================
__global__ void lat_kernel(const float* __restrict__ sums,
                           const float* __restrict__ counts,
                           const float* __restrict__ lattice,
                           const float* __restrict__ a2l_w,
                           const float* __restrict__ a2l_b,
                           const float* __restrict__ g,
                           const float* __restrict__ bb,
                           float* __restrict__ out) {
  int b = blockIdx.x;
  int t = threadIdx.x;  // 128 threads
  __shared__ float s9[9];
  if (t < 9) s9[t] = 0.0f;
  __syncthreads();
  float cnt = fmaxf(counts[b], 1.0f);
  float ai  = sums[b * 128 + t] / cnt;
#pragma unroll
  for (int j = 0; j < 9; ++j) atomicAdd(&s9[j], ai * a2l_w[t * 9 + j]);
  __syncthreads();
  if (t == 0) {
    float lf[9];
    float mu = 0.0f;
#pragma unroll
    for (int j = 0; j < 9; ++j) {
      lf[j] = lattice[b * 9 + j] + s9[j] + a2l_b[j];
      mu += lf[j];
    }
    mu *= (1.0f / 9.0f);
    float var = 0.0f;
#pragma unroll
    for (int j = 0; j < 9; ++j) { float d = lf[j] - mu; var += d * d; }
    var *= (1.0f / 9.0f);
    float rs = rsqrtf(var + 1e-5f);
#pragma unroll
    for (int j = 0; j < 9; ++j) out[b * 9 + j] = (lf[j] - mu) * rs * g[j] + bb[j];
  }
}

// ===========================================================================
// Host launcher
// ===========================================================================
extern "C" void kernel_launch(void* const* d_in, const int* in_sizes, int n_in,
                              void* d_out, int out_size, void* d_ws, size_t ws_size,
                              hipStream_t stream) {
  const float* x_in      = (const float*)d_in[0];
  const int*   edge_idx  = (const int*)d_in[1];
  const float* edge_attr = (const float*)d_in[2];
  const float* lattice   = (const float*)d_in[3];
  const int*   batch     = (const int*)d_in[4];
  const float* ee_w1 = (const float*)d_in[5],  *ee_b1 = (const float*)d_in[6];
  const float* ee_w2 = (const float*)d_in[7],  *ee_b2 = (const float*)d_in[8];
  const float* mn_w1 = (const float*)d_in[9],  *mn_b1 = (const float*)d_in[10];
  const float* mn_w2 = (const float*)d_in[11], *mn_b2 = (const float*)d_in[12];
  const float* un_w1 = (const float*)d_in[13], *un_b1 = (const float*)d_in[14];
  const float* un_w2 = (const float*)d_in[15], *un_b2 = (const float*)d_in[16];
  const float* l2a_w = (const float*)d_in[17], *l2a_b = (const float*)d_in[18];
  const float* a2l_w = (const float*)d_in[19], *a2l_b = (const float*)d_in[20];
  const float* nln_g = (const float*)d_in[21], *nln_b = (const float*)d_in[22];
  const float* lln_g = (const float*)d_in[23], *lln_b = (const float*)d_in[24];

  const int N  = in_sizes[0] / 128;
  const int E  = in_sizes[1] / 2;
  const int Bn = in_sizes[3] / 9;

  // workspace partition (256B aligned)
  char* ws = (char*)d_ws;
  size_t off = 0;
  auto alloc = [&](size_t bytes) {
    char* p = ws + off;
    off += (bytes + 255) & ~(size_t)255;
    return p;
  };
  _Float16* xh    = (_Float16*)alloc((size_t)N * 128 * 2);
  float* aggr     = (float*)alloc((size_t)N * 128 * 4);
  float* x2       = (float*)alloc((size_t)N * 128 * 4);
  float* latinfo  = (float*)alloc((size_t)Bn * 128 * 4);
  float* sums     = (float*)alloc((size_t)Bn * 128 * 4);
  float* counts   = (float*)alloc((size_t)Bn * 4);
  _Float16* wperm = (_Float16*)alloc((size_t)135168 * 2);

  float* out = (float*)d_out;

  // 1. weights -> f16, B-tile permuted, packed in ws (TDM source)
  wprep_kernel<<<(135168 + 255) / 256, 256, 0, stream>>>(
      ee_w1, ee_w2, mn_w1, mn_w2, un_w1, un_w2, wperm);
  // 2. x -> f16, zero aggr
  {
    int total = N * 128;
    cvt_kernel<<<(total + 255) / 256, 256, 0, stream>>>(x_in, xh, aggr, total);
  }
  // 3. lattice_info + zero sums/counts
  latinfo_kernel<<<(Bn * 128 + 255) / 256, 256, 0, stream>>>(
      lattice, l2a_w, l2a_b, latinfo, sums, counts, Bn);
  // 4. counts
  count_kernel<<<(N + 255) / 256, 256, 0, stream>>>(batch, counts, N);
  // 5. edge pipeline (WMMA + TDM), persistent blocks, 200KB dynamic LDS
  edge_kernel<<<512, 256, 204800, stream>>>(
      edge_attr, edge_idx, xh, wperm, ee_b1, ee_b2, mn_b1, mn_b2, aggr, E);
  // 6. node update (WMMA + TDM), 128KB dynamic LDS
  node_kernel<<<256, 256, 131072, stream>>>(
      x_in, xh, aggr, wperm + 86016, un_b1, un_b2, latinfo, batch, x2, sums, N);
  // 7. node layernorm -> out[0 : N*128)
  node_ln_kernel<<<(N + 7) / 8, 256, 0, stream>>>(x2, nln_g, nln_b, out, N);
  // 8. lattice path -> out[N*128 : N*128 + Bn*9)
  lat_kernel<<<Bn, 128, 0, stream>>>(sums, counts, lattice, a2l_w, a2l_b, lln_g,
                                     lln_b, out + (long long)N * 128);
}
